// DeepARDecoder_16003048145346
// MI455X (gfx1250) — matrix operand order repaired and checked by reading, not compiled, not run on hardware
//
#include <hip/hip_runtime.h>
#include <cstdint>
#include <cstddef>

// ---------------- problem constants ----------------
#define B_SZ   1024
#define H_SZ   1024
#define P_SZ   96
#define COV_SZ 31
#define S_SZ   100
#define K0     1056   // 32 (x) + 1024 (h0)
#define K1     2048   // 1024 (h0) + 1024 (h1)

// LDS staging geometry (per K-step of 32)
#define LDS_ROW   80            // 64B of K-data + 16B pad (bank de-phasing)
#define A_ROWS    64
#define B_ROWS    256           // 4 gates x 64 cols
#define A_BYTES   (A_ROWS * LDS_ROW)             // 5120
#define BUF_BYTES (A_BYTES + B_ROWS * LDS_ROW)   // 25600
#define CHUNKS    ((A_ROWS + B_ROWS) * 4)        // 1280 x 16B chunks
#define CHUNKS_PER_THREAD (CHUNKS / 256)         // 5

typedef __bf16 bf16;
typedef __attribute__((ext_vector_type(16))) __bf16 v16bf;
typedef __attribute__((ext_vector_type(8)))  __bf16 v8bf;
typedef __attribute__((ext_vector_type(8)))  float  v8f;
typedef int vint4 __attribute__((vector_size(16)));
typedef __attribute__((address_space(1))) vint4 g_vint4;
typedef __attribute__((address_space(3))) vint4 s_vint4;

__device__ __forceinline__ void sched_fence() {
#if __has_builtin(__builtin_amdgcn_sched_barrier)
  __builtin_amdgcn_sched_barrier(0);
#endif
}

// ---- async global->LDS copy (gfx1250 ASYNCcnt path) with safe fallback ----
__device__ __forceinline__ void copy16_async(const bf16* g, char* l) {
#if __has_builtin(__builtin_amdgcn_global_load_async_to_lds_b128)
  __builtin_amdgcn_global_load_async_to_lds_b128(
      (g_vint4*)g, (s_vint4*)l, 0, 0);
#else
  *(v8bf*)l = *(const v8bf*)g;
  *(v8bf*)(l + 16) = *(const v8bf*)(g + 8);
#endif
}
__device__ __forceinline__ void wait_async_le5() {
#if __has_builtin(__builtin_amdgcn_s_wait_asynccnt)
  __builtin_amdgcn_s_wait_asynccnt(5);
#else
  asm volatile("s_wait_asynccnt 0x5" ::: "memory");
#endif
}
__device__ __forceinline__ void wait_async_le0() {
#if __has_builtin(__builtin_amdgcn_s_wait_asynccnt)
  __builtin_amdgcn_s_wait_asynccnt(0);
#else
  asm volatile("s_wait_asynccnt 0x0" ::: "memory");
#endif
}

// ---- fragment loads from LDS (per CDNA5 ISA 7.12.2 layouts) ----
// A (16x32 bf16): lane l<16 row=l, elems0..7 = K hh*8.., elems8..15 = K 16+hh*8..
__device__ __forceinline__ v16bf lds_a_frag(const char* Ab, int row, int hh) {
  const char* p = Ab + row * LDS_ROW + hh * 16;
  v8bf lo = *(const v8bf*)p;
  v8bf hi = *(const v8bf*)(p + 32);
  v16bf r;
#pragma unroll
  for (int i = 0; i < 8; ++i) { r[i] = lo[i]; r[i + 8] = hi[i]; }
  return r;
}
// B (32x16 bf16): lane l<16 col=l, elems0..7 = K hh*8.., elems8..15 = K 16+hh*8..
__device__ __forceinline__ v16bf lds_b_frag(const char* Bb, int row, int hh) {
  const char* p = Bb + row * LDS_ROW + hh * 16;
  v8bf lo = *(const v8bf*)p;
  v8bf hi = *(const v8bf*)(p + 32);
  v16bf r;
#pragma unroll
  for (int i = 0; i < 8; ++i) { r[i] = lo[i]; r[i + 8] = hi[i]; }
  return r;
}

__device__ __forceinline__ float fsigmoid(float x) { return 1.0f / (1.0f + __expf(-x)); }
__device__ __forceinline__ float ftanh(float x)    { return 2.0f / (1.0f + __expf(-2.0f * x)) - 1.0f; }

// ---------------- one-time pack / init ----------------
__global__ void pack_kernel(
    const float* __restrict__ Wih0, const float* __restrict__ Whh0,
    const float* __restrict__ bih0, const float* __restrict__ bhh0,
    const float* __restrict__ Wih1, const float* __restrict__ Whh1,
    const float* __restrict__ bih1, const float* __restrict__ bhh1,
    const float* __restrict__ h_init, const float* __restrict__ c_init,
    const float* __restrict__ init_val,
    bf16* __restrict__ W0, bf16* __restrict__ W1,
    float* __restrict__ b0, float* __restrict__ b1,
    bf16* __restrict__ Z0a, bf16* __restrict__ Z1a,
    float* __restrict__ c0, float* __restrict__ c1, float* __restrict__ cur)
{
  size_t tid = (size_t)blockIdx.x * blockDim.x + threadIdx.x;
  size_t T   = (size_t)gridDim.x * blockDim.x;
  for (size_t i = tid; i < (size_t)4096 * K0; i += T) {
    size_t r = i / K0, k = i % K0;
    W0[i] = (bf16)(k < 32 ? Wih0[r * 32 + k] : Whh0[r * 1024 + (k - 32)]);
  }
  for (size_t i = tid; i < (size_t)4096 * K1; i += T) {
    size_t r = i / K1, k = i % K1;
    W1[i] = (bf16)(k < 1024 ? Wih1[r * 1024 + k] : Whh1[r * 1024 + (k - 1024)]);
  }
  for (size_t i = tid; i < 4096; i += T) { b0[i] = bih0[i] + bhh0[i]; b1[i] = bih1[i] + bhh1[i]; }
  for (size_t i = tid; i < (size_t)B_SZ * H_SZ; i += T) {
    size_t b = i / H_SZ, k = i % H_SZ;
    Z0a[b * K0 + 32 + k]   = (bf16)h_init[i];
    Z1a[b * K1 + 1024 + k] = (bf16)h_init[(size_t)B_SZ * H_SZ + i];
    c0[i] = c_init[i];
    c1[i] = c_init[(size_t)B_SZ * H_SZ + i];
  }
  for (size_t i = tid; i < B_SZ; i += T) cur[i] = init_val[i];
}

// ---------------- per-step: build x = [cur | cov_t] ----------------
__global__ void build_x_kernel(bf16* __restrict__ Z0, const float* __restrict__ cur,
                               const float* __restrict__ cov, int t)
{
  int b = blockIdx.x * blockDim.x + threadIdx.x;
  if (b >= B_SZ) return;
  bf16* row = Z0 + (size_t)b * K0;
  row[0] = (bf16)cur[b];
  const float* c = cov + ((size_t)b * P_SZ + t) * COV_SZ;
#pragma unroll
  for (int k = 0; k < COV_SZ; ++k) row[1 + k] = (bf16)c[k];
}

// ------------ fused GEMM + LSTM pointwise (one layer) -------------
// gates = Z (B x K) @ W^T (4H x K) + bias; block tile: 64 batch x 64 hidden x 4 gates.
// K-loop: double-buffered async global->LDS staging; per iteration: preload A + all 8
// B fragments from LDS (hard sched fence), then 8 back-to-back v_wmma_f32_16x16x32_bf16.
__global__ __launch_bounds__(256) void lstm_cell_kernel(
    const bf16* __restrict__ Z, int K,
    const bf16* __restrict__ W, const float* __restrict__ bias,
    float* __restrict__ C,
    bf16* __restrict__ hA, int sA, int cA,
    bf16* __restrict__ hB, int sB, int cB,
    float* __restrict__ hF)
{
  __shared__ __attribute__((aligned(16))) char smem[2 * BUF_BYTES];

  const int tid  = threadIdx.x;
  const int lane = tid & 31;
  const int l16  = lane & 15;
  const int hh   = lane >> 4;
  const int wave = tid >> 5;
  const int wm   = wave & 3;
  const int wn   = wave >> 2;
  const int mBlock = blockIdx.x * 64;
  const int jBlock = blockIdx.y * 64;
  const int mBase  = mBlock + wm * 16;

  // precompute this thread's 5 copy chunks (row/col mapping is kb-invariant)
  const bf16* gsrc[CHUNKS_PER_THREAD];
  int         loff[CHUNKS_PER_THREAD];
#pragma unroll
  for (int j = 0; j < CHUNKS_PER_THREAD; ++j) {
    int c = tid + j * 256;
    if (c < A_ROWS * 4) {
      int row = c >> 2, kq = c & 3;
      gsrc[j] = Z + (size_t)(mBlock + row) * K + kq * 8;
      loff[j] = row * LDS_ROW + kq * 16;
    } else {
      int cc = c - A_ROWS * 4;
      int row = cc >> 2, kq = cc & 3;
      int wRow = (row >> 6) * H_SZ + jBlock + (row & 63);
      gsrc[j] = W + (size_t)wRow * K + kq * 8;
      loff[j] = A_BYTES + row * LDS_ROW + kq * 16;
    }
  }

  v8f zero = {};
  v8f acc[4][2];
#pragma unroll
  for (int g = 0; g < 4; ++g)
#pragma unroll
    for (int ns = 0; ns < 2; ++ns) acc[g][ns] = zero;

  const int aRowL = wm * 16 + l16;
  const int nIter = K / 32;

  // compute one staged K-slice: preload frags (fenced), then back-to-back WMMAs
  auto compute_slice = [&](int buf) {
    const char* Ab = smem + buf * BUF_BYTES;
    const char* Bb = Ab + A_BYTES;
    v16bf afrag = lds_a_frag(Ab, aRowL, hh);
    v16bf bfrag[4][2];
#pragma unroll
    for (int g = 0; g < 4; ++g)
#pragma unroll
      for (int ns = 0; ns < 2; ++ns)
        bfrag[g][ns] = lds_b_frag(Bb, g * 64 + wn * 32 + ns * 16 + l16, hh);
    sched_fence();   // keep all ds_loads before, all WMMAs after
#pragma unroll
    for (int g = 0; g < 4; ++g)
#pragma unroll
      for (int ns = 0; ns < 2; ++ns)
        acc[g][ns] = __builtin_amdgcn_wmma_f32_16x16x32_bf16(
            false, afrag, false, bfrag[g][ns], (short)0, acc[g][ns], false, false);
  };

  // prologue: stage K-slice 0 into buffer 0
#pragma unroll
  for (int j = 0; j < CHUNKS_PER_THREAD; ++j)
    copy16_async(gsrc[j], smem + loff[j]);

  // steady state: always issue next slice, then consume current
  for (int it = 0; it < nIter - 1; ++it) {
    const int cur = it & 1;
    const int nxt = cur ^ 1;
    __syncthreads();                       // buffer 'nxt' no longer being read
    const int koff = (it + 1) * 32;        // elements
#pragma unroll
    for (int j = 0; j < CHUNKS_PER_THREAD; ++j)
      copy16_async(gsrc[j] + koff, smem + nxt * BUF_BYTES + loff[j]);
    wait_async_le5();                      // batch 'it' landed (in-order completion)
    __syncthreads();                       // everyone sees buffer 'cur'
    compute_slice(cur);
  }
  // tail iteration
  __syncthreads();
  wait_async_le0();
  __syncthreads();
  compute_slice((nIter - 1) & 1);

  // D layout: VGPR v, lane: n = lane&15, m = v + 8*(lane>>4)
#pragma unroll
  for (int ns = 0; ns < 2; ++ns) {
#pragma unroll
    for (int v = 0; v < 8; ++v) {
      int m = mBase + v + 8 * hh;
      int n = jBlock + wn * 32 + ns * 16 + l16;
      float gi = acc[0][ns][v] + bias[n];
      float gf = acc[1][ns][v] + bias[H_SZ + n];
      float gg = acc[2][ns][v] + bias[2 * H_SZ + n];
      float go = acc[3][ns][v] + bias[3 * H_SZ + n];
      float si = fsigmoid(gi);
      float sf = fsigmoid(gf);
      float so = fsigmoid(go);
      float tg = ftanh(gg);
      size_t idx = (size_t)m * H_SZ + n;
      float cN = sf * C[idx] + si * tg;
      C[idx] = cN;
      float h = so * ftanh(cN);
      if (hA) hA[(size_t)m * sA + cA + n] = (bf16)h;
      if (hB) hB[(size_t)m * sB + cB + n] = (bf16)h;
      if (hF) hF[idx] = h;
    }
  }
}

// ------------- projection + softplus + sampling -------------
__global__ __launch_bounds__(256) void proj_sample_kernel(
    const float* __restrict__ h1, const float* __restrict__ Wp, const float* __restrict__ bp,
    const float* __restrict__ noise, int t,
    float* __restrict__ outP, float* __restrict__ outS, float* __restrict__ cur)
{
  int lane = threadIdx.x & 31;
  int wave = threadIdx.x >> 5;
  int b = blockIdx.x * 8 + wave;
  float s0 = 0.0f, s1 = 0.0f;
  const float* hr = h1 + (size_t)b * H_SZ;
  for (int k = lane; k < H_SZ; k += 32) {
    float hv = hr[k];
    s0 += hv * Wp[k];
    s1 += hv * Wp[H_SZ + k];
  }
#pragma unroll
  for (int off = 16; off > 0; off >>= 1) {
    s0 += __shfl_down(s0, off, 32);
    s1 += __shfl_down(s1, off, 32);
  }
  float mu  = __shfl(s0, 0, 32) + bp[0];
  float pre = __shfl(s1, 0, 32) + bp[1];
  float sp  = pre > 20.0f ? pre : log1pf(__expf(pre));
  float sigma = sp + 1e-6f;
  if (lane == 0) {
    outP[((size_t)b * P_SZ + t) * 2 + 0] = mu;
    outP[((size_t)b * P_SZ + t) * 2 + 1] = sigma;
  }
  for (int s = lane; s < S_SZ; s += 32) {
    float val = mu + sigma * noise[((size_t)t * S_SZ + s) * B_SZ + b];
    outS[((size_t)s * B_SZ + b) * P_SZ + t] = val;   // (S, B, P)
    if (s == 0) cur[b] = val;
  }
}

// ---------------- host-side orchestration ----------------
extern "C" void kernel_launch(void* const* d_in, const int* in_sizes, int n_in,
                              void* d_out, int out_size, void* d_ws, size_t ws_size,
                              hipStream_t stream) {
  (void)in_sizes; (void)n_in; (void)out_size; (void)ws_size;
  const float* cov    = (const float*)d_in[0];
  const float* iv     = (const float*)d_in[1];
  const float* noise  = (const float*)d_in[2];
  const float* h_init = (const float*)d_in[3];
  const float* c_init = (const float*)d_in[4];
  const float* Wih0   = (const float*)d_in[5];
  const float* Whh0   = (const float*)d_in[6];
  const float* bih0   = (const float*)d_in[7];
  const float* bhh0   = (const float*)d_in[8];
  const float* Wih1   = (const float*)d_in[9];
  const float* Whh1   = (const float*)d_in[10];
  const float* bih1   = (const float*)d_in[11];
  const float* bhh1   = (const float*)d_in[12];
  const float* Wp     = (const float*)d_in[13];
  const float* bp     = (const float*)d_in[14];

  float* outP = (float*)d_out;                                  // (B, P, 2)
  float* outS = outP + (size_t)B_SZ * P_SZ * 2;                 // (S, B, P)

  char* ws = (char*)d_ws;
  size_t off = 0;
  auto carve = [&](size_t bytes) -> void* {
    void* p = ws + off; off += (bytes + 255) & ~(size_t)255; return p;
  };
  bf16*  W0    = (bf16*)carve((size_t)4096 * K0 * 2);
  bf16*  W1    = (bf16*)carve((size_t)4096 * K1 * 2);
  float* b0    = (float*)carve(4096 * 4);
  float* b1    = (float*)carve(4096 * 4);
  bf16*  Z0a   = (bf16*)carve((size_t)B_SZ * K0 * 2);
  bf16*  Z0b   = (bf16*)carve((size_t)B_SZ * K0 * 2);
  bf16*  Z1a   = (bf16*)carve((size_t)B_SZ * K1 * 2);
  bf16*  Z1b   = (bf16*)carve((size_t)B_SZ * K1 * 2);
  float* c0    = (float*)carve((size_t)B_SZ * H_SZ * 4);
  float* c1    = (float*)carve((size_t)B_SZ * H_SZ * 4);
  float* h1f   = (float*)carve((size_t)B_SZ * H_SZ * 4);
  float* cur   = (float*)carve(B_SZ * 4);

  pack_kernel<<<2048, 256, 0, stream>>>(Wih0, Whh0, bih0, bhh0, Wih1, Whh1, bih1, bhh1,
                                        h_init, c_init, iv,
                                        W0, W1, b0, b1, Z0a, Z1a, c0, c1, cur);

  dim3 gemmGrid(B_SZ / 64, H_SZ / 64);   // 16 x 16 workgroups
  for (int t = 0; t < P_SZ; ++t) {
    bf16* Z0cur  = (t & 1) ? Z0b : Z0a;
    bf16* Z0next = (t & 1) ? Z0a : Z0b;
    bf16* Z1cur  = (t & 1) ? Z1b : Z1a;
    bf16* Z1next = (t & 1) ? Z1a : Z1b;

    build_x_kernel<<<4, 256, 0, stream>>>(Z0cur, cur, cov, t);

    // layer 0: h0 -> next step's Z0 (cols 32..) and this step's Z1 (cols 0..)
    lstm_cell_kernel<<<gemmGrid, 256, 0, stream>>>(
        Z0cur, K0, W0, b0, c0,
        Z0next, K0, 32,
        Z1cur,  K1, 0,
        nullptr);

    // layer 1: h1 -> next step's Z1 (cols 1024..) + fp32 copy for projection
    lstm_cell_kernel<<<gemmGrid, 256, 0, stream>>>(
        Z1cur, K1, W1, b1, c1,
        Z1next, K1, 1024,
        nullptr, 0, 0,
        h1f);

    proj_sample_kernel<<<B_SZ / 8, 256, 0, stream>>>(h1f, Wp, bp, noise, t, outP, outS, cur);
  }
}
